// BidirectionalEdgeGraphNetwork_13305808683263
// MI455X (gfx1250) — compile-verified, matrix operand order checked by reading, not run on hardware
//
#include <hip/hip_runtime.h>
#include <hip/hip_bf16.h>

// ---------------------------------------------------------------------------
// Types for CDNA5 WMMA (wave32): D(16x16 f32) = A(16x32 f16) * B(32x16 f16) + C
// ---------------------------------------------------------------------------
typedef __attribute__((ext_vector_type(16))) _Float16 v16h;
typedef __attribute__((ext_vector_type(8)))  float    v8f;

union H16 { _Float16 h[16]; v16h v; };

__device__ inline v16h cvt16(const float* __restrict__ p) {
    H16 t;
#pragma unroll
    for (int i = 0; i < 16; ++i) t.h[i] = (_Float16)p[i];
    return t.v;
}

// Ordered-uint encoding of float for atomic max (monotonic map)
__device__ inline unsigned fenc(float f) {
    unsigned u = __float_as_uint(f);
    return (u & 0x80000000u) ? ~u : (u | 0x80000000u);
}
__device__ inline float fdec(unsigned k) {
    unsigned u = (k & 0x80000000u) ? (k & 0x7FFFFFFFu) : ~k;
    return __uint_as_float(u);
}
#define ENC_NEGINF 0x007FFFFFu   // fenc(-inf)

// ---------------------------------------------------------------------------
// A-matrix fetch functors (row m, column k) -> v16h of 16 consecutive K values
// ---------------------------------------------------------------------------
struct AfPlainF16 {
    const _Float16* A; int lda;
    __device__ v16h load(int m, int k) const { return *(const v16h*)(A + (size_t)m * lda + k); }
};
// concat [ x[row[e]] (128) | ef[e] (128) | ef[pair(e)] (128) | x[col[e]] (128) ], f16 sources
struct AfEdgeConcatH {
    const _Float16* x; const _Float16* ef; const int* row; const int* col; int half;
    __device__ v16h load(int e, int k) const {
        const _Float16* p;
        if (k < 128)       p = x  + (size_t)row[e] * 128 + k;
        else if (k < 256)  p = ef + (size_t)e      * 128 + (k - 128);
        else if (k < 384) {
            int re = (e < half) ? (e + half) : (e - half);
            p = ef + (size_t)re * 128 + (k - 256);
        } else             p = x  + (size_t)col[e] * 128 + (k - 384);
        return *(const v16h*)p;
    }
};
// concat of two N x 128 f32 matrices (small node-side GEMMs)
struct AfConcat2 {
    const float* a; const float* b;
    __device__ v16h load(int m, int k) const {
        const float* p = (k < 128) ? (a + (size_t)m * 128 + k)
                                   : (b + (size_t)m * 128 + (k - 128));
        return cvt16(p);
    }
};

// ---------------------------------------------------------------------------
// Weight packer: f32 row-major W[K][Nw] -> f16 WMMA-B tiles.
// tile index = nt*Kt + kt ; within tile: lane l holds column n = nt*16+(l&15),
// K values kt*32 + (l>>4)*16 + j , j = 0..15 (contiguous v16h per lane).
// ---------------------------------------------------------------------------
__global__ void __launch_bounds__(256)
pack_w_kernel(const float* __restrict__ W, _Float16* __restrict__ P, int K, int Nw) {
    int Kt = K >> 5, Nt = Nw >> 4;
    int tile = blockIdx.x * (blockDim.x >> 5) + (threadIdx.x >> 5);
    if (tile >= Kt * Nt) return;
    int nt = tile / Kt, kt = tile - nt * Kt;
    int lane = threadIdx.x & 31;
    int n  = (nt << 4) + (lane & 15);
    int kb = (kt << 5) + ((lane >> 4) << 4);
    H16 t;
#pragma unroll
    for (int j = 0; j < 16; ++j) t.h[j] = (_Float16)W[(size_t)(kb + j) * Nw + n];
    *(v16h*)(P + ((size_t)tile * 32 + lane) * 16) = t.v;
}

// f32 -> f16 bulk convert
__global__ void __launch_bounds__(256)
cvt_f16_kernel(const float* __restrict__ src, _Float16* __restrict__ dst, long long n) {
    long long i = (long long)blockIdx.x * blockDim.x + threadIdx.x;
    if (i < n) dst[i] = (_Float16)src[i];
}

// ---------------------------------------------------------------------------
// Register-blocked WMMA GEMM.
// Block = 256 threads (8 waves). Wave w computes M-tile (blockIdx.x*8+w) x
// NTW consecutive N-tiles (panel blockIdx.y), holding NTW f32 accumulators.
// The B panel (NTW*Ktiles WMMA tiles) is staged in LDS once per block via
// CDNA5 async global->LDS copies (ASYNCcnt) and shared by all 8 waves.
// act: 0 = none, 1 = relu, 2 = sigmoid.
// ---------------------------------------------------------------------------
template <int NTW, typename AF>
__global__ void __launch_bounds__(256)
gemm_wmma(AF af, const _Float16* __restrict__ Bp, const float* __restrict__ bias,
          float* __restrict__ Cf32, _Float16* __restrict__ Cf16, int ldc,
          int Mtiles, int Ktiles, int act) {
    extern __shared__ _Float16 Bs[];          // NTW * Ktiles * 512 f16
    int panel = blockIdx.y;

    // ---- stage B panel: global -> LDS, asynchronous (ASYNCcnt) ------------
    const char* src = (const char*)(Bp + (size_t)panel * NTW * Ktiles * 512);
    int n16 = NTW * Ktiles * 64;              // 16-byte chunks in panel
    for (int i = threadIdx.x; i < n16; i += blockDim.x) {
        unsigned lds_off = (unsigned)i * 16u; // dynamic-LDS base is offset 0
        const void* g = src + (size_t)i * 16;
        asm volatile("global_load_async_to_lds_b128 %0, %1, off"
                     :: "v"(lds_off), "v"(g) : "memory");
    }
    asm volatile("s_wait_asynccnt 0x0" ::: "memory");
    __syncthreads();

    int wave = threadIdx.x >> 5, lane = threadIdx.x & 31;
    int mt = blockIdx.x * 8 + wave;
    if (mt >= Mtiles) return;

    int mA   = mt * 16 + (lane & 15);
    int kOff = (lane >> 4) << 4;
    v8f acc[NTW];
#pragma unroll
    for (int j = 0; j < NTW; ++j) acc[j] = (v8f){0.f, 0.f, 0.f, 0.f, 0.f, 0.f, 0.f, 0.f};

    for (int kt = 0; kt < Ktiles; ++kt) {
        v16h a = af.load(mA, (kt << 5) + kOff);
#pragma unroll
        for (int j = 0; j < NTW; ++j) {
            v16h b = *(const v16h*)(Bs + (((j * Ktiles + kt) * 32 + lane) << 4));
            acc[j] = __builtin_amdgcn_wmma_f32_16x16x32_f16(false, a, false, b,
                                                            (short)0, acc[j], false, false);
        }
    }

    int mHi = (lane >> 4) ? 8 : 0;
#pragma unroll
    for (int j = 0; j < NTW; ++j) {
        int n  = (panel * NTW + j) * 16 + (lane & 15);
        float bv = bias ? bias[n] : 0.f;
#pragma unroll
        for (int r = 0; r < 8; ++r) {
            float v = acc[j][r] + bv;
            if (act == 1) v = fmaxf(v, 0.f);
            else if (act == 2) v = 1.f / (1.f + __expf(-v));
            int m = mt * 16 + r + mHi;
            if (Cf32) Cf32[(size_t)m * ldc + n] = v;
            if (Cf16) Cf16[(size_t)m * ldc + n] = (_Float16)v;
        }
    }
}

// ---------------------------------------------------------------------------
// Distance-gate MLP per edge: 4 -> 32 (relu) -> 1 (sigmoid)
// ---------------------------------------------------------------------------
__global__ void __launch_bounds__(256)
dm_kernel(const float* __restrict__ desc, const int* __restrict__ row,
          const int* __restrict__ col, const float* __restrict__ dW1,
          const float* __restrict__ dB1, const float* __restrict__ dW2,
          const float* __restrict__ dB2, float* __restrict__ dm, int E) {
    int e = blockIdx.x * blockDim.x + threadIdx.x;
    if (e >= E) return;
    int r = row[e], c = col[e];
    float d0 = desc[(size_t)r * 8 + 0] - desc[(size_t)c * 8 + 0];
    float d1 = desc[(size_t)r * 8 + 1] - desc[(size_t)c * 8 + 1];
    float d2 = desc[(size_t)r * 8 + 2] - desc[(size_t)c * 8 + 2];
    float f[4] = {d0, d1, d2, sqrtf(d0 * d0 + d1 * d1 + d2 * d2)};
    float acc = dB2[0];
#pragma unroll 4
    for (int j = 0; j < 32; ++j) {
        float s = dB1[j];
#pragma unroll
        for (int i = 0; i < 4; ++i) s += f[i] * dW1[i * 32 + j];
        acc += fmaxf(s, 0.f) * dW2[j];
    }
    dm[e] = 1.f / (1.f + __expf(-acc));
}

// ---------------------------------------------------------------------------
// Per-(edge, head) attention: two tiny matvecs, softmax over 16, weight v,
// emit prob to output and ordered-uint atomic-max into agg.
// ---------------------------------------------------------------------------
__global__ void __launch_bounds__(256)
attention_kernel(const float* __restrict__ xq, const float* __restrict__ xv,
                 const float* __restrict__ kef, const float* __restrict__ dm,
                 const int* __restrict__ row, const int* __restrict__ col,
                 const float* __restrict__ attW1, const float* __restrict__ attB1,
                 const float* __restrict__ attW2, const float* __restrict__ attB2,
                 float* __restrict__ probs, unsigned* __restrict__ aggkey, int E) {
    __shared__ float sW1[1024], sW2[512], sB1[32], sB2[16];
    for (int i = threadIdx.x; i < 1024; i += blockDim.x) sW1[i] = attW1[i];
    for (int i = threadIdx.x; i < 512;  i += blockDim.x) sW2[i] = attW2[i];
    if (threadIdx.x < 32) sB1[threadIdx.x] = attB1[threadIdx.x];
    if (threadIdx.x < 16) sB2[threadIdx.x] = attB2[threadIdx.x];
    __syncthreads();

    int t = blockIdx.x * blockDim.x + threadIdx.x;
    if (t >= E * 8) return;
    int e = t >> 3, hd = t & 7;
    int r = row[e], c = col[e];
    __builtin_prefetch(xv + (size_t)c * 128, 0, 1);

    float hv[32];
#pragma unroll
    for (int i = 0; i < 16; ++i) hv[i]      = xq[(size_t)r * 128 + i * 8 + hd];
#pragma unroll
    for (int i = 0; i < 16; ++i) hv[16 + i] = kef[(size_t)e * 128 + i * 8 + hd];

    float h1[32];
#pragma unroll 4
    for (int o = 0; o < 32; ++o) {
        float s = sB1[o];
#pragma unroll
        for (int cc = 0; cc < 32; ++cc) s += sW1[o * 32 + cc] * hv[cc];
        h1[o] = fmaxf(s, 0.f);
    }
    float g = dm[e];
    float att[16], mx = -1e30f;
#pragma unroll
    for (int o = 0; o < 16; ++o) {
        float s = sB2[o];
#pragma unroll
        for (int cc = 0; cc < 32; ++cc) s += sW2[o * 32 + cc] * h1[cc];
        att[o] = s * g;
        mx = fmaxf(mx, att[o]);
    }
    float sum = 0.f;
#pragma unroll
    for (int o = 0; o < 16; ++o) { att[o] = __expf((att[o] - mx) * 0.25f); sum += att[o]; }
    float inv = 1.f / sum;
#pragma unroll
    for (int o = 0; o < 16; ++o) {
        float pr = att[o] * inv;
        probs[(size_t)e * 128 + o * 8 + hd] = pr;
        float w = pr * xv[(size_t)c * 128 + o * 8 + hd];
        atomicMax(&aggkey[(size_t)r * 128 + o * 8 + hd], fenc(w));
    }
}

// ---------------------------------------------------------------------------
// Small utility kernels
// ---------------------------------------------------------------------------
__global__ void fill_u32(unsigned* __restrict__ p, unsigned v, int n) {
    int i = blockIdx.x * blockDim.x + threadIdx.x;
    if (i < n) p[i] = v;
}
__global__ void degree_kernel(const int* __restrict__ row, const int* __restrict__ col,
                              float* __restrict__ dout, float* __restrict__ din, int E) {
    int e = blockIdx.x * blockDim.x + threadIdx.x;
    if (e >= E) return;
    atomicAdd(dout + row[e], 1.f);
    atomicAdd(din  + col[e], 1.f);
}
__global__ void segsum_kernel(const float* __restrict__ upd, const int* __restrict__ row,
                              const int* __restrict__ col, float* __restrict__ outf,
                              float* __restrict__ incf, int E) {
    int idx = blockIdx.x * blockDim.x + threadIdx.x;
    if (idx >= E * 128) return;
    int e = idx >> 7, j = idx & 127;
    float v = upd[idx];
    atomicAdd(outf + (size_t)row[e] * 128 + j, v);
    atomicAdd(incf + (size_t)col[e] * 128 + j, v);
}
__global__ void normalize_kernel(float* __restrict__ outf, float* __restrict__ incf,
                                 const float* __restrict__ dout, const float* __restrict__ din,
                                 const unsigned* __restrict__ aggkey, float* __restrict__ agg,
                                 int N) {
    int idx = blockIdx.x * blockDim.x + threadIdx.x;
    if (idx >= N * 128) return;
    int n = idx >> 7;
    outf[idx] /= fmaxf(dout[n], 1.f);
    incf[idx] /= fmaxf(din[n], 1.f);
    unsigned k = aggkey[idx];
    agg[idx] = (k <= ENC_NEGINF) ? 0.f : fdec(k);
}
__global__ void final_node_kernel(const float* __restrict__ updn, const float* __restrict__ eatt,
                                  float* __restrict__ xout, int n, int relu) {
    int i = blockIdx.x * blockDim.x + threadIdx.x;
    if (i >= n) return;
    float v = fmaxf(updn[i], 0.f) * eatt[i];
    if (relu) v = fmaxf(v, 0.f);
    xout[i] = v;
}
__global__ void final_edge_kernel(const float* __restrict__ upde, float* __restrict__ efout,
                                  long long n, int relu) {
    long long i = (long long)blockIdx.x * blockDim.x + threadIdx.x;
    if (i >= n) return;
    float v = upde[i];
    if (relu) v = fmaxf(v, 0.f);
    efout[i] = v;
}

// ---------------------------------------------------------------------------
// Host-side orchestration
// ---------------------------------------------------------------------------
extern "C" void kernel_launch(void* const* d_in, const int* in_sizes, int n_in,
                              void* d_out, int out_size, void* d_ws, size_t ws_size,
                              hipStream_t stream) {
    const int N = 10000, E = 160000, HALF = 80000, L = 2;

    const float* x0   = (const float*)d_in[0];
    const float* ef0  = (const float*)d_in[1];
    const int*   edges= (const int*)d_in[2];
    const float* desc = (const float*)d_in[3];
    const float* qW = (const float*)d_in[4],  * qB = (const float*)d_in[5];
    const float* kW = (const float*)d_in[6],  * kB = (const float*)d_in[7];
    const float* vW = (const float*)d_in[8],  * vB = (const float*)d_in[9];
    const float* dW1= (const float*)d_in[10], * dB1= (const float*)d_in[11];
    const float* dW2= (const float*)d_in[12], * dB2= (const float*)d_in[13];
    const float* eW1= (const float*)d_in[14], * eB1= (const float*)d_in[15];
    const float* eW2= (const float*)d_in[16], * eB2= (const float*)d_in[17];
    const float* aW1= (const float*)d_in[18], * aB1= (const float*)d_in[19];
    const float* aW2= (const float*)d_in[20], * aB2= (const float*)d_in[21];
    const float* nW1= (const float*)d_in[22], * nB1= (const float*)d_in[23];
    const float* nW2= (const float*)d_in[24], * nB2= (const float*)d_in[25];
    const float* aW = (const float*)d_in[26], * aB = (const float*)d_in[27];

    const int* row = edges;
    const int* col = edges + E;

    // workspace bump allocator (256 B aligned)
    char* ws = (char*)d_ws;
    size_t off = 0;
    auto alloc = [&](size_t bytes) -> char* {
        off = (off + 255) & ~(size_t)255;
        char* p = ws + off;
        off += bytes;
        return p;
    };

    _Float16 *pq[2], *pv[2], *pk[2], *pe1[2], *pe2[2], *pn1[2], *pn2[2], *pa[2];
    for (int l = 0; l < L; ++l) {
        pq[l]  = (_Float16*)alloc(16384 * 2);
        pv[l]  = (_Float16*)alloc(16384 * 2);
        pk[l]  = (_Float16*)alloc(16384 * 2);
        pe1[l] = (_Float16*)alloc(196608 * 2);
        pe2[l] = (_Float16*)alloc(49152 * 2);
        pn1[l] = (_Float16*)alloc(65536 * 2);
        pn2[l] = (_Float16*)alloc(32768 * 2);
        pa[l]  = (_Float16*)alloc(32768 * 2);
    }
    float* xq   = (float*)alloc((size_t)N * 128 * 4);
    float* xv   = (float*)alloc((size_t)N * 128 * 4);
    float* kef  = (float*)alloc((size_t)E * 128 * 4);
    float* dm   = (float*)alloc((size_t)E * 4);
    _Float16* xh  = (_Float16*)alloc((size_t)N * 128 * 2);
    _Float16* efh = (_Float16*)alloc((size_t)E * 128 * 2);
    _Float16* hide = (_Float16*)alloc((size_t)E * 384 * 2);
    float* upde = (float*)alloc((size_t)E * 128 * 4);
    unsigned* aggkey = (unsigned*)alloc((size_t)N * 128 * 4);
    float* agg  = (float*)alloc((size_t)N * 128 * 4);
    float* outf = (float*)alloc((size_t)N * 128 * 4);
    float* incf = (float*)alloc((size_t)N * 128 * 4);
    float* doutg= (float*)alloc((size_t)N * 4);
    float* ding = (float*)alloc((size_t)N * 4);
    _Float16* hidn = (_Float16*)alloc((size_t)N * 256 * 2);
    float* updn = (float*)alloc((size_t)N * 128 * 4);
    float* eatt = (float*)alloc((size_t)N * 128 * 4);
    float* xcur = (float*)alloc((size_t)N * 128 * 4);
    float* efcur= (float*)alloc((size_t)E * 128 * 4);

    auto packW = [&](const float* W, _Float16* P, int K, int Nw) {
        int tiles = (K >> 5) * (Nw >> 4);
        pack_w_kernel<<<(tiles + 7) / 8, 256, 0, stream>>>(W, P, K, Nw);
    };
    for (int l = 0; l < L; ++l) {
        packW(qW  + (size_t)l * 128 * 128, pq[l],  128, 128);
        packW(vW  + (size_t)l * 128 * 128, pv[l],  128, 128);
        packW(kW  + (size_t)l * 128 * 128, pk[l],  128, 128);
        packW(eW1 + (size_t)l * 512 * 384, pe1[l], 512, 384);
        packW(eW2 + (size_t)l * 384 * 128, pe2[l], 384, 128);
        packW(nW1 + (size_t)l * 256 * 256, pn1[l], 256, 256);
        packW(nW2 + (size_t)l * 256 * 128, pn2[l], 256, 128);
        packW(aW  + (size_t)l * 256 * 128, pa[l],  256, 128);
    }

    float* out_x  = (float*)d_out;
    float* out_ef = out_x + (size_t)N * 128;
    float* out_pr = out_ef + (size_t)E * 128;

    const int MT_N = N / 16;   // 625 node row-tiles
    const int MT_E = E / 16;   // 10000 edge row-tiles

    for (int l = 0; l < L; ++l) {
        const float* xin  = (l == 0) ? x0  : xcur;
        const float* efin = (l == 0) ? ef0 : efcur;
        int relu = (l < L - 1) ? 1 : 0;

        // ---- f16 copies of activations for WMMA A operands -----------------
        cvt_f16_kernel<<<(N * 128 + 255) / 256, 256, 0, stream>>>(xin, xh, (long long)N * 128);
        cvt_f16_kernel<<<((long long)E * 128 + 255) / 256, 256, 0, stream>>>(efin, efh, (long long)E * 128);

        // ---- per-node q / v projections, per-edge k projection -------------
        {
            AfPlainF16 af{xh, 128};
            gemm_wmma<8><<<dim3((MT_N + 7) / 8, 1), 256, 8 * 4 * 1024, stream>>>(
                af, pq[l], qB + (size_t)l * 128, xq, (_Float16*)nullptr, 128, MT_N, 4, 0);
            gemm_wmma<8><<<dim3((MT_N + 7) / 8, 1), 256, 8 * 4 * 1024, stream>>>(
                af, pv[l], vB + (size_t)l * 128, xv, (_Float16*)nullptr, 128, MT_N, 4, 0);
        }
        {
            AfPlainF16 af{efh, 128};
            gemm_wmma<8><<<dim3((MT_E + 7) / 8, 1), 256, 8 * 4 * 1024, stream>>>(
                af, pk[l], kB + (size_t)l * 128, kef, (_Float16*)nullptr, 128, MT_E, 4, 0);
        }

        // ---- distance gate -------------------------------------------------
        dm_kernel<<<(E + 255) / 256, 256, 0, stream>>>(
            desc, row, col, dW1 + (size_t)l * 128, dB1 + (size_t)l * 32,
            dW2 + (size_t)l * 32, dB2 + (size_t)l * 1, dm, E);

        // ---- edge MLP: concat(512) -> 384 relu -> 128 ----------------------
        {
            AfEdgeConcatH af{xh, efh, row, col, HALF};
            gemm_wmma<8><<<dim3((MT_E + 7) / 8, 3), 256, 8 * 16 * 1024, stream>>>(
                af, pe1[l], eB1 + (size_t)l * 384, (float*)nullptr, hide, 384, MT_E, 16, 1);
        }
        {
            AfPlainF16 af{hide, 384};
            gemm_wmma<8><<<dim3((MT_E + 7) / 8, 1), 256, 8 * 12 * 1024, stream>>>(
                af, pe2[l], eB2 + (size_t)l * 128, upde, (_Float16*)nullptr, 128, MT_E, 12, 0);
        }

        // ---- reductions init ----------------------------------------------
        fill_u32<<<(N * 128 + 255) / 256, 256, 0, stream>>>(aggkey, ENC_NEGINF, N * 128);
        hipMemsetAsync(outf, 0, (size_t)N * 128 * 4, stream);
        hipMemsetAsync(incf, 0, (size_t)N * 128 * 4, stream);
        hipMemsetAsync(doutg, 0, (size_t)N * 4, stream);
        hipMemsetAsync(ding,  0, (size_t)N * 4, stream);

        // ---- attention + segment-max --------------------------------------
        attention_kernel<<<(E * 8 + 255) / 256, 256, 0, stream>>>(
            xq, xv, kef, dm, row, col,
            aW1 + (size_t)l * 1024, aB1 + (size_t)l * 32,
            aW2 + (size_t)l * 512,  aB2 + (size_t)l * 16,
            out_pr + (size_t)l * E * 128, aggkey, E);

        // ---- degrees + segment sums + normalize ---------------------------
        degree_kernel<<<(E + 255) / 256, 256, 0, stream>>>(row, col, doutg, ding, E);
        segsum_kernel<<<(E * 128 + 255) / 256, 256, 0, stream>>>(upde, row, col, outf, incf, E);
        normalize_kernel<<<(N * 128 + 255) / 256, 256, 0, stream>>>(
            outf, incf, doutg, ding, aggkey, agg, N);

        // ---- node MLP: concat(x, agg)(256) -> 256 relu -> 128 --------------
        {
            AfConcat2 af{xin, agg};
            gemm_wmma<8><<<dim3((MT_N + 7) / 8, 2), 256, 8 * 8 * 1024, stream>>>(
                af, pn1[l], nB1 + (size_t)l * 256, (float*)nullptr, hidn, 256, MT_N, 8, 1);
        }
        {
            AfPlainF16 af{hidn, 256};
            gemm_wmma<8><<<dim3((MT_N + 7) / 8, 1), 256, 8 * 8 * 1024, stream>>>(
                af, pn2[l], nB2 + (size_t)l * 128, updn, (_Float16*)nullptr, 128, MT_N, 8, 0);
        }
        // ---- edge-attention gate: concat(outf, incf)(256) -> 128 sigmoid ---
        {
            AfConcat2 af{outf, incf};
            gemm_wmma<8><<<dim3((MT_N + 7) / 8, 1), 256, 8 * 8 * 1024, stream>>>(
                af, pa[l], aB + (size_t)l * 128, eatt, (_Float16*)nullptr, 128, MT_N, 8, 2);
        }

        // ---- combine + inter-layer relu -----------------------------------
        float* xdst  = (l < L - 1) ? xcur  : out_x;
        float* efdst = (l < L - 1) ? efcur : out_ef;
        final_node_kernel<<<(N * 128 + 255) / 256, 256, 0, stream>>>(
            updn, eatt, xdst, N * 128, relu);
        final_edge_kernel<<<((long long)E * 128 + 255) / 256, 256, 0, stream>>>(
            upde, efdst, (long long)E * 128, relu);
    }
}
